// TinyOnnMoE_62723702390906
// MI455X (gfx1250) — matrix operand order repaired and checked by reading, not compiled, not run on hardware
//
#include <hip/hip_runtime.h>
#include <hip/hip_bf16.h>
#include <math.h>

// ---------------------------------------------------------------------------
// MoE (top-2 of 8 experts, SwiGLU) for MI455X / gfx1250, wave32 + WMMA bf16,
// async Global<->LDS (ASYNCcnt) data movement for bf16 tiles and epilogues.
// Shapes fixed by the reference: B=2, S=2048, H=768, F=1536, E=8, k=2.
// ---------------------------------------------------------------------------

typedef __bf16 bf16;
typedef __bf16 bf16x2  __attribute__((ext_vector_type(2)));
typedef __bf16 bf16x8  __attribute__((ext_vector_type(8)));
typedef __bf16 bf16x16 __attribute__((ext_vector_type(16)));
typedef float  f32x8   __attribute__((ext_vector_type(8)));

static constexpr int Tt = 4096;   // tokens = B*S
static constexpr int H  = 768;
static constexpr int F  = 1536;
static constexpr int E  = 8;

static constexpr int BM   = 64;   // block tile M (tokens)
static constexpr int BN   = 128;  // block tile N
static constexpr int KT   = 64;   // K step = 2 WMMA K-slices
static constexpr int LPAD = 8;    // LDS row pad -> conflict-free ds_load_b128

#define WMMA_BF16(a, b, c) \
  __builtin_amdgcn_wmma_f32_16x16x32_bf16(false, (a), false, (b), (short)0, (c), false, false)

static __device__ __forceinline__ bf16x16 join8(bf16x8 lo, bf16x8 hi) {
  return __builtin_shufflevector(lo, hi, 0,1,2,3,4,5,6,7,8,9,10,11,12,13,14,15);
}

static __device__ __forceinline__ unsigned pack2(float a, float b) {
  bf16x2 v; v[0] = (bf16)a; v[1] = (bf16)b;      // -> v_cvt_pk_bf16_f32
  return __builtin_bit_cast(unsigned, v);
}

// ---- CDNA5 async Global<->LDS (ASYNCcnt) ----------------------------------
static __device__ __forceinline__ void async_load_b128(const void* g, const void* l) {
  unsigned lo = (unsigned)(uintptr_t)l;          // LDS byte offset (low 32 bits)
  asm volatile("global_load_async_to_lds_b128 %0, %1, off"
               :: "v"(lo), "v"(g) : "memory");
}
static __device__ __forceinline__ void async_store_b128(void* g, const void* l) {
  unsigned lo = (unsigned)(uintptr_t)l;
  asm volatile("global_store_async_from_lds_b128 %0, %1, off"
               :: "v"(g), "v"(lo) : "memory");
}
static __device__ __forceinline__ void wait_async0() {
  asm volatile("s_wait_asynccnt 0x0" ::: "memory");
}

// ---------------------------------------------------------------------------
// Router: fp32 logits -> softmax -> top-2 -> renormalized weights; builds
// per-expert gather lists of encoded rows enc = token*2 + slot.
// ---------------------------------------------------------------------------
__global__ void moe_router(const float* __restrict__ x, const float* __restrict__ gw,
                           float* __restrict__ logits, float* __restrict__ tokw,
                           int* __restrict__ lists, int* __restrict__ cnt) {
  __shared__ float sl[32][E];
  const int lt = threadIdx.x >> 3;
  const int e  = threadIdx.x & 7;
  const int t  = blockIdx.x * 32 + lt;

  const float* xr = x  + (size_t)t * H;
  const float* gr = gw + (size_t)e * H;
  float acc = 0.f;
  #pragma unroll 4
  for (int h = 0; h < H; h += 4) {
    acc += xr[h] * gr[h] + xr[h+1] * gr[h+1] + xr[h+2] * gr[h+2] + xr[h+3] * gr[h+3];
  }
  logits[(size_t)t * E + e] = acc;
  sl[lt][e] = acc;
  __syncthreads();

  if (threadIdx.x < 32) {
    const int tt = blockIdx.x * 32 + threadIdx.x;
    float p[E];
    float m = sl[threadIdx.x][0];
    #pragma unroll
    for (int i = 1; i < E; i++) m = fmaxf(m, sl[threadIdx.x][i]);
    #pragma unroll
    for (int i = 0; i < E; i++) p[i] = __expf(sl[threadIdx.x][i] - m);
    int i1 = 0;
    #pragma unroll
    for (int i = 1; i < E; i++) if (p[i] > p[i1]) i1 = i;     // first index on ties
    int i2 = (i1 == 0) ? 1 : 0;
    #pragma unroll
    for (int i = 0; i < E; i++) if (i != i1 && p[i] > p[i2]) i2 = i;
    const float s = p[i1] + p[i2];
    tokw[tt * 2 + 0] = p[i1] / s;
    tokw[tt * 2 + 1] = p[i2] / s;
    int pos = atomicAdd(&cnt[i1], 1);
    lists[i1 * Tt + pos] = tt * 2 + 0;
    pos = atomicAdd(&cnt[i2], 1);
    lists[i2 * Tt + pos] = tt * 2 + 1;
  }
}

// ---------------------------------------------------------------------------
// GEMM1: gathered rows: g = x*w1^T, u = x*w3^T (shared A fragments),
// act[enc][F] = silu(g)*u in bf16. 8 waves x (2x2) WMMA subtiles.
// ---------------------------------------------------------------------------
__global__ __launch_bounds__(256)
void moe_gemm1(const float* __restrict__ x, const float* __restrict__ w1,
               const float* __restrict__ w3, const int* __restrict__ lists,
               const int* __restrict__ cnt, bf16* __restrict__ act) {
  const int e  = blockIdx.z;
  const int ne = cnt[e];
  const int m0 = blockIdx.x * BM;
  if (m0 >= ne) return;
  const int n0 = blockIdx.y * BN;

  __shared__ __align__(16) bf16 lA [BM][KT + LPAD];
  __shared__ __align__(16) bf16 lB1[BN][KT + LPAD];
  __shared__ __align__(16) bf16 lB3[BN][KT + LPAD];
  __shared__ __align__(16) bf16 lO [BM][BN + LPAD];
  __shared__ int lenc[BM];

  const int tid = threadIdx.x;
  if (tid < BM) lenc[tid] = (m0 + tid < ne) ? lists[e * Tt + m0 + tid] : -1;
  __syncthreads();

  // load roles: A = 4 thr/row * 16 floats, B = 2 thr/row * 32 floats
  const int ar = tid >> 2, ac = (tid & 3) * 16;
  const int br = tid >> 1, bc = (tid & 1) * 32;
  const int encA = lenc[ar];
  const float* xr  = (encA >= 0) ? (x + (size_t)(encA >> 1) * H + ac) : nullptr;
  const float* w1r = w1 + ((size_t)e * F + (n0 + br)) * H + bc;
  const float* w3r = w3 + ((size_t)e * F + (n0 + br)) * H + bc;

  f32x8 accG[2][2] = {};
  f32x8 accU[2][2] = {};

  const int wave = tid >> 5, lane = tid & 31;
  const int wm = (wave & 1) * 32, wn = (wave >> 1) * 32;
  const int lhalf = (lane >= 16) ? 1 : 0;
  const int lidx  = lane & 15;

  for (int k0 = 0; k0 < H; k0 += KT) {
    // ---- stage tiles (fp32 -> bf16 via v_cvt_pk_bf16_f32) ----
    {
      unsigned* dA = (unsigned*)&lA[ar][ac];
      if (xr) {
        #pragma unroll
        for (int j = 0; j < 8; j++) dA[j] = pack2(xr[k0 + 2*j], xr[k0 + 2*j + 1]);
      } else {
        #pragma unroll
        for (int j = 0; j < 8; j++) dA[j] = 0u;
      }
      unsigned* d1 = (unsigned*)&lB1[br][bc];
      unsigned* d3 = (unsigned*)&lB3[br][bc];
      #pragma unroll
      for (int j = 0; j < 16; j++) d1[j] = pack2(w1r[k0 + 2*j], w1r[k0 + 2*j + 1]);
      #pragma unroll
      for (int j = 0; j < 16; j++) d3[j] = pack2(w3r[k0 + 2*j], w3r[k0 + 2*j + 1]);
      if (k0 + KT < H) {
        __builtin_prefetch(w1r + k0 + KT, 0, 1);
        __builtin_prefetch(w3r + k0 + KT, 0, 1);
      }
    }
    __syncthreads();

    #pragma unroll
    for (int kk = 0; kk < KT; kk += 32) {
      bf16x16 aF[2], b1F[2], b3F[2];
      #pragma unroll
      for (int mi = 0; mi < 2; mi++) {
        const bf16* ap = &lA[wm + mi * 16 + lidx][kk + lhalf * 8];
        aF[mi] = join8(*(const bf16x8*)ap, *(const bf16x8*)(ap + 16));
      }
      #pragma unroll
      for (int ni = 0; ni < 2; ni++) {
        const bf16* p1 = &lB1[wn + ni * 16 + lidx][kk + lhalf * 8];
        b1F[ni] = join8(*(const bf16x8*)p1, *(const bf16x8*)(p1 + 16));
        const bf16* p3 = &lB3[wn + ni * 16 + lidx][kk + lhalf * 8];
        b3F[ni] = join8(*(const bf16x8*)p3, *(const bf16x8*)(p3 + 16));
      }
      #pragma unroll
      for (int mi = 0; mi < 2; mi++)
        #pragma unroll
        for (int ni = 0; ni < 2; ni++) {
          accG[mi][ni] = WMMA_BF16(aF[mi], b1F[ni], accG[mi][ni]);
          accU[mi][ni] = WMMA_BF16(aF[mi], b3F[ni], accU[mi][ni]);
        }
    }
    __syncthreads();
  }

  // ---- epilogue: silu(g)*u -> LDS -> async LDS->global scatter by enc ----
  #pragma unroll
  for (int mi = 0; mi < 2; mi++)
    #pragma unroll
    for (int ni = 0; ni < 2; ni++)
      #pragma unroll
      for (int i = 0; i < 8; i++) {
        const float g = accG[mi][ni][i];
        const float u = accU[mi][ni][i];
        const float a = (g / (1.f + __expf(-g))) * u;
        lO[wm + mi * 16 + i + lhalf * 8][wn + ni * 16 + lidx] = (bf16)a;
      }
  __syncthreads();
  {
    const int r = tid >> 2, c = (tid & 3) * 32;
    const int enc = lenc[r];
    if (enc >= 0) {
      bf16* dst = act + (size_t)enc * F + n0 + c;
      #pragma unroll
      for (int j = 0; j < 4; j++)
        async_store_b128(dst + 8 * j, &lO[r][c + 8 * j]);
    }
  }
  wait_async0();   // (s_endpgm also waits, explicit for clarity)
}

// ---------------------------------------------------------------------------
// GEMM2: partial[enc][H] = act[enc] * w2[e]^T (fp32; combine weight later).
// A tile is bf16 in global -> async gather straight into LDS.
// ---------------------------------------------------------------------------
__global__ __launch_bounds__(256)
void moe_gemm2(const bf16* __restrict__ act, const float* __restrict__ w2,
               const int* __restrict__ lists, const int* __restrict__ cnt,
               float* __restrict__ partial) {
  const int e  = blockIdx.z;
  const int ne = cnt[e];
  const int m0 = blockIdx.x * BM;
  if (m0 >= ne) return;
  const int n0 = blockIdx.y * BN;

  __shared__ __align__(16) bf16  lA[BM][KT + LPAD];
  __shared__ __align__(16) bf16  lB[BN][KT + LPAD];
  __shared__ __align__(16) float lO[BM][BN + 4];
  __shared__ int lenc[BM];

  const int tid = threadIdx.x;
  if (tid < BM) lenc[tid] = (m0 + tid < ne) ? lists[e * Tt + m0 + tid] : -1;
  __syncthreads();

  const int ar = tid >> 2, ac = (tid & 3) * 16;   // 16 bf16 (32B) per thread
  const int br = tid >> 1, bc = (tid & 1) * 32;
  const int encA = lenc[ar];
  const bf16*  actr = (encA >= 0) ? (act + (size_t)encA * F + ac) : nullptr;
  const float* w2r  = w2 + ((size_t)e * H + (n0 + br)) * F + bc;

  f32x8 acc[2][2] = {};

  const int wave = tid >> 5, lane = tid & 31;
  const int wm = (wave & 1) * 32, wn = (wave >> 1) * 32;
  const int lhalf = (lane >= 16) ? 1 : 0;
  const int lidx  = lane & 15;

  for (int k0 = 0; k0 < F; k0 += KT) {
    // ---- A: async global->LDS gather (bf16, no VGPR round trip) ----
    if (actr) {
      async_load_b128(actr + k0,     &lA[ar][ac]);
      async_load_b128(actr + k0 + 8, &lA[ar][ac + 8]);
    } else {
      uint4* z = (uint4*)&lA[ar][ac];
      z[0] = make_uint4(0u, 0u, 0u, 0u);
      z[1] = make_uint4(0u, 0u, 0u, 0u);
    }
    // ---- B: fp32 -> bf16 convert-stage ----
    {
      unsigned* dB = (unsigned*)&lB[br][bc];
      #pragma unroll
      for (int j = 0; j < 16; j++) dB[j] = pack2(w2r[k0 + 2*j], w2r[k0 + 2*j + 1]);
      if (k0 + KT < F) __builtin_prefetch(w2r + k0 + KT, 0, 1);
    }
    wait_async0();
    __syncthreads();

    #pragma unroll
    for (int kk = 0; kk < KT; kk += 32) {
      bf16x16 aF[2], bF[2];
      #pragma unroll
      for (int mi = 0; mi < 2; mi++) {
        const bf16* ap = &lA[wm + mi * 16 + lidx][kk + lhalf * 8];
        aF[mi] = join8(*(const bf16x8*)ap, *(const bf16x8*)(ap + 16));
      }
      #pragma unroll
      for (int ni = 0; ni < 2; ni++) {
        const bf16* bp = &lB[wn + ni * 16 + lidx][kk + lhalf * 8];
        bF[ni] = join8(*(const bf16x8*)bp, *(const bf16x8*)(bp + 16));
      }
      #pragma unroll
      for (int mi = 0; mi < 2; mi++)
        #pragma unroll
        for (int ni = 0; ni < 2; ni++)
          acc[mi][ni] = WMMA_BF16(aF[mi], bF[ni], acc[mi][ni]);
    }
    __syncthreads();
  }

  #pragma unroll
  for (int mi = 0; mi < 2; mi++)
    #pragma unroll
    for (int ni = 0; ni < 2; ni++)
      #pragma unroll
      for (int i = 0; i < 8; i++)
        lO[wm + mi * 16 + i + lhalf * 8][wn + ni * 16 + lidx] = acc[mi][ni][i];
  __syncthreads();
  {
    const int r = tid >> 2, c = (tid & 3) * 32;
    const int enc = lenc[r];
    if (enc >= 0) {
      float* dst = partial + (size_t)enc * H + n0 + c;
      #pragma unroll
      for (int j = 0; j < 8; j++)
        async_store_b128(dst + 4 * j, &lO[r][c + 4 * j]);
    }
  }
  wait_async0();
}

// ---------------------------------------------------------------------------
// Combine: y[t] = w[t,0]*partial[2t] + w[t,1]*partial[2t+1]  (deterministic)
// ---------------------------------------------------------------------------
__global__ void moe_combine(const float* __restrict__ partial,
                            const float* __restrict__ tokw,
                            float* __restrict__ y) {
  const int idx = blockIdx.x * blockDim.x + threadIdx.x;
  if (idx >= Tt * H) return;
  const int t = idx / H, h = idx % H;
  y[idx] = tokw[2 * t]     * partial[(size_t)(2 * t)     * H + h] +
           tokw[2 * t + 1] * partial[(size_t)(2 * t + 1) * H + h];
}

// ---------------------------------------------------------------------------
extern "C" void kernel_launch(void* const* d_in, const int* in_sizes, int n_in,
                              void* d_out, int out_size, void* d_ws, size_t ws_size,
                              hipStream_t stream) {
  (void)in_sizes; (void)n_in; (void)out_size; (void)ws_size;
  const float* x  = (const float*)d_in[0];   // [T, H]
  const float* gw = (const float*)d_in[1];   // [E, H]
  const float* w1 = (const float*)d_in[2];   // [E, F, H]
  const float* w3 = (const float*)d_in[3];   // [E, F, H]
  const float* w2 = (const float*)d_in[4];   // [E, H, F]

  float* y      = (float*)d_out;             // [T, H]
  float* logits = y + (size_t)Tt * H;        // [T, E]

  // workspace layout (~50.5 MB)
  char* ws = (char*)d_ws;
  bf16*  act     = (bf16*)ws;                                       // 2T*F bf16
  float* partial = (float*)(ws + (size_t)2 * Tt * F * 2);           // 2T*H fp32
  float* tokw    = (float*)((char*)partial + (size_t)2 * Tt * H * 4);
  int*   lists   = (int*)((char*)tokw + (size_t)2 * Tt * 4);        // E*T ints
  int*   cnt     = (int*)((char*)lists + (size_t)E * Tt * 4);       // E ints

  hipMemsetAsync(cnt, 0, E * sizeof(int), stream);

  moe_router<<<Tt / 32, 256, 0, stream>>>(x, gw, logits, tokw, lists, cnt);

  dim3 g1(Tt / BM, F / BN, E);   // 64 x 12 x 8
  moe_gemm1<<<g1, 256, 0, stream>>>(x, w1, w3, lists, cnt, act);

  dim3 g2(Tt / BM, H / BN, E);   // 64 x 6 x 8
  moe_gemm2<<<g2, 256, 0, stream>>>(act, w2, lists, cnt, partial);

  moe_combine<<<(Tt * H) / 256, 256, 0, stream>>>(partial, tokw, y);
}
